// input_attention_layer_33225867001899
// MI455X (gfx1250) — matrix operand order repaired and checked by reading, not compiled, not run on hardware
//
#include <hip/hip_runtime.h>

// CDNA5 / gfx1250 vector types for WMMA
typedef __attribute__((ext_vector_type(16))) _Float16 v16h;
typedef __attribute__((ext_vector_type(2)))  _Float16 v2h;
typedef __attribute__((ext_vector_type(8)))  float    v8f;
typedef __attribute__((ext_vector_type(2)))  float    v2f;

static constexpr int kP = 64;            // feature dim
static constexpr int kJ = 64;            // window dim
static constexpr int kTokPerBlk = 16;    // tokens per block (one per wave32)

// Fast sigmoid: v_exp_f32 + v_rcp_f32 (no IEEE div expansion)
__device__ __forceinline__ float fast_sigmoid(float x) {
    return __builtin_amdgcn_rcpf(1.f + __expf(-x));
}

// Block = 512 threads = 16 wave32; wave w owns token blockIdx*16 + w.
// Waves 0..3 additionally perform the 16x64x64 GEMM (t1 @ w2) via WMMA f16.
__global__ __launch_bounds__(512)
void input_attention_kernel(const float* __restrict__ x,
                            const float* __restrict__ w1,
                            const float* __restrict__ b1,
                            const float* __restrict__ w2,
                            const float* __restrict__ b2,
                            float* __restrict__ out_a,
                            float* __restrict__ out_attn)
{
    __shared__ float    w1s[64];
    __shared__ float    b1s[64];
    __shared__ float    b2s[64];
    __shared__ _Float16 w2t[64][66];   // w2 transposed: [p][j], padded (66) for banks
    __shared__ _Float16 t1h[16][72];   // t1 f16 per token, padded stride 72 (conflict-free A loads)
    __shared__ float    t2s[16][65];   // t2 then attn, padded

    const int tid  = threadIdx.x;
    const int wave = tid >> 5;         // 0..15 == token within block
    const int lane = tid & 31;
    const int hs   = lane >> 4;        // half-wave select (0/1)
    const int l15  = lane & 15;
    const size_t tok = (size_t)blockIdx.x * kTokPerBlk + wave;

    // ---------------- Phase 0: stage parameters into LDS ----------------
    if (tid < 64) {
        w1s[tid] = w1[tid];
        b1s[tid] = b1[tid];
        b2s[tid] = b2[tid];
    }
#pragma unroll
    for (int i = 0; i < 8; ++i) {      // 512 threads x 8 = 4096 elems of w2
        const int e = tid + 512 * i;   // coalesced read of w2[j][p] (p fast)
        const int j = e >> 6;
        const int p = e & 63;
        w2t[p][j] = (_Float16)w2[e];   // store transposed as f16
    }
    __syncthreads();

    // ---------------- Phase 1: t1 = sigmoid(w1 . x_tok + b1) ----------------
    // Lane l handles columns j = 2l, 2l+1. Wave reads 256B/row, coalesced, RT
    // cache policy so the phase-4 re-read of this 16KB tile stays L2-hot.
    const float* xt = x + tok * (size_t)(kP * kJ);
    float a0 = 0.f, a1 = 0.f;
#pragma unroll 8
    for (int p = 0; p < kP; ++p) {
        const v2f v = *(const v2f*)(xt + p * kJ + 2 * lane);
        const float wv = w1s[p];
        a0 = fmaf(wv, v.x, a0);
        a1 = fmaf(wv, v.y, a1);
    }
    a0 = fast_sigmoid(a0 + b1s[2 * lane]);
    a1 = fast_sigmoid(a1 + b1s[2 * lane + 1]);
    {
        v2h t1p;
        t1p.x = (_Float16)a0;
        t1p.y = (_Float16)a1;
        *(v2h*)&t1h[wave][2 * lane] = t1p;
    }
    __syncthreads();

    // ---------------- Phase 2: t2 = sigmoid(t1 @ w2 + b2) via WMMA ----------------
    // 16(tokens) x 64(K=j) x 64(N=p) GEMM; wave n in [0,4) computes N-tile n.
    // Wave-uniform branch keeps EXEC all-1s as WMMA requires.
    if (wave < 4) {
        const int n = wave;
        v8f acc = {};
#pragma unroll
        for (int kk = 0; kk < 2; ++kk) {           // K halves: j in [0,32), [32,64)
            v16h A, Bf;
#pragma unroll
            for (int v = 0; v < 8; ++v) {
                // A 16x32 f16 layout (ISA 7.12.2): VGPR v, lane half hs:
                //   j = 32*kk + (v<4 ? 0 : 16) + 8*hs + 2*(v&3) (+0/+1 packed)
                const int ja = 32 * kk + ((v & 4) << 2) + 8 * hs + 2 * (v & 3);
                const v2h pa = *(const v2h*)&t1h[l15][ja];
                A[2 * v]     = pa.x;
                A[2 * v + 1] = pa.y;
                // B 32x16 f16 layout: lane carries N=l15; same K order as A
                const int jb = 32 * kk + ((v & 4) << 2) + 8 * hs + 2 * (v & 3);
                const v2h pb = *(const v2h*)&w2t[16 * n + l15][jb];
                Bf[2 * v]     = pb.x;
                Bf[2 * v + 1] = pb.y;
            }
            acc = __builtin_amdgcn_wmma_f32_16x16x32_f16(
                /*neg_a=*/false, A, /*neg_b=*/false, Bf,
                /*c_mod=*/(short)0, acc, /*reuse_a=*/false, /*reuse_b=*/false);
        }
        // D layout: VGPR r -> token M = r + 8*hs, column p = 16*n + l15
        const float bb = b2s[16 * n + l15];
#pragma unroll
        for (int r = 0; r < 8; ++r) {
            t2s[r + 8 * hs][16 * n + l15] = fast_sigmoid(acc[r] + bb);
        }
    }
    __syncthreads();

    // ---------------- Phase 3: softmax over p (64 vals, 2 per lane) ----------------
    float v0 = t2s[wave][2 * lane];
    float v1 = t2s[wave][2 * lane + 1];
    float m = fmaxf(v0, v1);
#pragma unroll
    for (int off = 16; off >= 1; off >>= 1) m = fmaxf(m, __shfl_xor(m, off, 32));
    float e0 = __expf(v0 - m);
    float e1 = __expf(v1 - m);
    float s = e0 + e1;
#pragma unroll
    for (int off = 16; off >= 1; off >>= 1) s += __shfl_xor(s, off, 32);
    const float inv = __builtin_amdgcn_rcpf(s);
    e0 *= inv;
    e1 *= inv;
    t2s[wave][2 * lane]     = e0;   // attn back into LDS for phase 4 broadcast
    t2s[wave][2 * lane + 1] = e1;
    {
        v2f at;
        at.x = e0;
        at.y = e1;
        __builtin_nontemporal_store(at, (v2f*)(out_attn + tok * kP + 2 * lane));
    }
    __syncthreads();

    // ---------------- Phase 4: a[j] = sum_p attn[p] * x[p][j] ----------------
    // Second pass over the token tile: L2-hot from phase 1; nontemporal so we
    // don't re-rinse dead lines back through the hierarchy.
    float c0 = 0.f, c1 = 0.f;
#pragma unroll 8
    for (int p = 0; p < kP; ++p) {
        const v2f v = __builtin_nontemporal_load((const v2f*)(xt + p * kJ + 2 * lane));
        const float ap = t2s[wave][p];   // LDS broadcast, conflict-free
        c0 = fmaf(ap, v.x, c0);
        c1 = fmaf(ap, v.y, c1);
    }
    {
        v2f av;
        av.x = c0;
        av.y = c1;
        __builtin_nontemporal_store(av, (v2f*)(out_a + tok * kJ + 2 * lane));
    }
}

extern "C" void kernel_launch(void* const* d_in, const int* in_sizes, int n_in,
                              void* d_out, int out_size, void* d_ws, size_t ws_size,
                              hipStream_t stream) {
    const float* x  = (const float*)d_in[0];  // [B,T,P,J] fp32
    const float* w1 = (const float*)d_in[1];  // [1,P]
    const float* b1 = (const float*)d_in[2];  // [1,J]
    const float* w2 = (const float*)d_in[3];  // [J,P]
    const float* b2 = (const float*)d_in[4];  // [1,P]
    float* out = (float*)d_out;               // a [B,T,J] then attn [B,T,P]

    const int nTok = in_sizes[0] / (kP * kJ); // 16384
    float* out_a    = out;
    float* out_attn = out + (size_t)nTok * kJ;

    dim3 grid(nTok / kTokPerBlk);
    dim3 block(512);
    hipLaunchKernelGGL(input_attention_kernel, grid, block, 0, stream,
                       x, w1, b1, w2, b2, out_a, out_attn);
}
